// ProteinMPNN_77970836291591
// MI455X (gfx1250) — matrix operand order repaired and checked by reading, not compiled, not run on hardware
//
#include <hip/hip_runtime.h>
#include <cstdint>

#define DEVINL __device__ __forceinline__

typedef __attribute__((ext_vector_type(16))) _Float16 v16h;
typedef __attribute__((ext_vector_type(8)))  float    v8f;

constexpr int BB = 2;
constexpr int NN = 512;
constexpr int NK = 48;
constexpr int DH = 128;
constexpr float SCALE_ = 30.0f;
constexpr float LN_EPS_ = 1e-5f;

DEVINL float geluf(float x){ return 0.5f*x*(1.0f + erff(x*0.70710678118654752f)); }

// ---- WMMA fragment loaders (layouts per CDNA5 ISA 7.12.2) ----
// A: 16x32 f16, lane<16: M=lane, K = kb+{0..7} (v0..3) and kb+16+{0..7} (v4..7)
//    lane>=16: same M, K-halves +8.
DEVINL v16h load_a_frag(const _Float16* A, int astride, int mtile, int lane, int kb){
  const _Float16* row = A + (mtile*16 + (lane&15))*astride + kb + ((lane>>4)<<3);
  v16h a;
#pragma unroll
  for (int j=0;j<8;++j) a[j]   = row[j];
#pragma unroll
  for (int j=0;j<8;++j) a[8+j] = row[16+j];
  return a;
}
// B: 32x16 f16 from transposed weights Wt[N][K]; lane<16: col=lane, K=kb+{0..7} and kb+16+{0..7};
//    lane>=16: col=lane-16, K-halves +8.
DEVINL v16h load_b_frag(const _Float16* Wt, int Kdim, int ntile, int lane, int kb){
  const _Float16* col = Wt + (size_t)(ntile*16 + (lane&15))*Kdim + kb + ((lane>>4)<<3);
  v16h b;
#pragma unroll
  for (int j=0;j<16;++j) b[j] = col[(j&7) + ((j>>3)<<4)];
  return b;
}

DEVINL v8f gemm_tile(const _Float16* A, int astride, const _Float16* Wt, int Kdim,
                     int mtile, int ntile, int lane){
  v8f acc = {};
  for (int kb=0; kb<Kdim; kb+=32){
    v16h a = load_a_frag(A, astride, mtile, lane, kb);
    v16h b = load_b_frag(Wt, Kdim, ntile, lane, kb);
    acc = __builtin_amdgcn_wmma_f32_16x16x32_f16(false, a, false, b, (short)0, acc, false, false);
  }
  return acc;
}

// C/D layout: lane -> col = lane&15; row = mtile*16 + 8*(lane>>4) + v
DEVINL void store_act_gelu(_Float16* Ao, int ostride, int mtile, int ntile, int lane,
                           v8f acc, const float* bias){
  int n  = ntile*16 + (lane&15);
  int m0 = mtile*16 + ((lane>>4)<<3);
  float bv = bias[n];
#pragma unroll
  for (int v=0; v<8; ++v) Ao[(m0+v)*ostride + n] = (_Float16)geluf(acc[v] + bv);
}

DEVINL float blk_sum128(float v, float* red){
  int tid = threadIdx.x;
  red[tid] = v; __syncthreads();
#pragma unroll
  for (int s=64; s>0; s>>=1){ if (tid<s) red[tid] += red[tid+s]; __syncthreads(); }
  float r = red[0]; __syncthreads();
  return r;
}

// ---------------- utility kernels ----------------
__global__ void k_w2h(const float* src, _Float16* dst, int Kd, int Nd){
  int idx = blockIdx.x*256 + threadIdx.x;
  if (idx < Kd*Nd){
    int n = idx / Kd, k = idx - n*Kd;
    dst[idx] = (_Float16)src[(size_t)k*Nd + n];
  }
}

__global__ void k_inv(const int* dorder, int* invp){
  int idx = blockIdx.x*256 + threadIdx.x;
  if (idx < BB*NN){
    int b = idx / NN;
    invp[b*NN + dorder[idx]] = idx - b*NN;
  }
}

__global__ __launch_bounds__(256) void k_knn(const float* ca, const float* mask, int* eidx){
  __shared__ float dadj[NN];
  __shared__ float rv[256];
  __shared__ int   ri[256];
  int bi = blockIdx.x, b = bi/NN, tid = threadIdx.x;
  float xi = ca[(size_t)bi*3+0], yi = ca[(size_t)bi*3+1], zi = ca[(size_t)bi*3+2];
  float mi = mask[bi];
  float lmax = -1e30f;
  for (int j=tid; j<NN; j+=256){
    float mj = mask[b*NN+j];
    float m2 = mi*mj;
    float dx = xi - ca[((size_t)b*NN+j)*3+0];
    float dy = yi - ca[((size_t)b*NN+j)*3+1];
    float dz = zi - ca[((size_t)b*NN+j)*3+2];
    float d = m2 * sqrtf(dx*dx+dy*dy+dz*dz + 1e-6f);
    dadj[j] = d;
    lmax = fmaxf(lmax, d);
  }
  rv[tid] = lmax; __syncthreads();
  for (int s=128; s>0; s>>=1){ if (tid<s) rv[tid]=fmaxf(rv[tid],rv[tid+s]); __syncthreads(); }
  float Dmax = rv[0]; __syncthreads();
  for (int j=tid; j<NN; j+=256){
    float mj = mask[b*NN+j]; float m2 = mi*mj;
    dadj[j] = dadj[j] + (1.0f-m2)*Dmax;
  }
  __syncthreads();
  for (int s=0; s<NK; ++s){
    float bv = 1e30f; int bj = 1<<30;
    for (int j=tid; j<NN; j+=256){
      float v = dadj[j];
      if (v < bv || (v == bv && j < bj)){ bv = v; bj = j; }
    }
    rv[tid]=bv; ri[tid]=bj; __syncthreads();
    for (int st=128; st>0; st>>=1){
      if (tid<st){
        float ov=rv[tid+st]; int oj=ri[tid+st];
        if (ov<rv[tid] || (ov==rv[tid] && oj<ri[tid])){ rv[tid]=ov; ri[tid]=oj; }
      }
      __syncthreads();
    }
    if (tid==0){ eidx[(size_t)bi*NK+s] = ri[0]; dadj[ri[0]] = 1e30f; }
    __syncthreads();
  }
}

__global__ __launch_bounds__(128) void k_node_embed(const float* nf, const float* g,
                                                    const float* bt, const float* W,
                                                    const float* bias, float* hV){
  __shared__ float x[128];
  __shared__ float red[128];
  int bi = blockIdx.x, tid = threadIdx.x;
  float v = nf[(size_t)bi*DH + tid];
  float mu = blk_sum128(v, red) * (1.0f/128.f);
  float d = v - mu;
  float var = blk_sum128(d*d, red) * (1.0f/128.f);
  x[tid] = g[tid]*d*rsqrtf(var+LN_EPS_) + bt[tid];
  __syncthreads();
  float acc = bias[tid];
  for (int k=0; k<128; ++k) acc += x[k]*W[(size_t)k*128 + tid];
  hV[(size_t)bi*DH + tid] = acc;
}

__global__ __launch_bounds__(128) void k_edge_embed(const float* pf, const int* eidx,
                                                    const float* g, const float* bt,
                                                    const _Float16* Wt, const float* bias,
                                                    float* hE){
  constexpr int AS = 136;
  __shared__ _Float16 A[NK*AS];
  __shared__ int nbS[NK];
  int bi = blockIdx.x, tid = threadIdx.x, lane = tid&31, w = tid>>5;
  if (tid < NK) nbS[tid] = eidx[(size_t)bi*NK + tid];
  __syncthreads();
  for (int r = w*12; r < w*12+12; ++r){
    const float* src = pf + ((size_t)bi*NN + nbS[r])*DH;
    float vv[4]; float s = 0.f;
#pragma unroll
    for (int q=0;q<4;++q){ vv[q]=src[lane+32*q]; s += vv[q]; }
    for (int m=16;m>0;m>>=1) s += __shfl_xor(s,m,32);
    float mu = s*(1.f/128.f), s2 = 0.f;
#pragma unroll
    for (int q=0;q<4;++q){ float d=vv[q]-mu; s2 += d*d; }
    for (int m=16;m>0;m>>=1) s2 += __shfl_xor(s2,m,32);
    float rin = rsqrtf(s2*(1.f/128.f)+LN_EPS_);
#pragma unroll
    for (int q=0;q<4;++q){ int c=lane+32*q; A[r*AS+c] = (_Float16)(g[c]*(vv[q]-mu)*rin + bt[c]); }
  }
  __syncthreads();
  for (int nt=2*w; nt<2*w+2; ++nt)
    for (int mt=0; mt<3; ++mt){
      v8f acc = gemm_tile(A, AS, Wt, 128, mt, nt, lane);
      int n = nt*16 + (lane&15), m0 = mt*16 + ((lane>>4)<<3);
      float bv = bias[n];
#pragma unroll
      for (int v=0; v<8; ++v) hE[((size_t)bi*NK + m0+v)*DH + n] = acc[v] + bv;
    }
}

// ---------------- encoder node update ----------------
struct EncNodeArgs {
  const float* hV_in; float* hV_out;
  const float* hE; const int* eidx; const float* mask;
  const _Float16 *W1t,*W2t,*W3t;
  const float *b1,*b2,*b3;
  const float *din,*dinb,*dout,*doutb;
  const float *n1g,*n1b,*n2g,*n2b;
  const float *a_node;
};

__global__ __launch_bounds__(128) void k_enc_node(EncNodeArgs A){
  constexpr int AS1 = 400, AS2 = 144;
  __shared__ _Float16 A1[NK*AS1];
  __shared__ _Float16 A2[NK*AS2];
  __shared__ int   nbS[NK];
  __shared__ float mattS[NK];
  __shared__ float hv[128];
  int bi = blockIdx.x, tid = threadIdx.x, lane = tid&31, w = tid>>5;
  int b = bi/NN;
  float mi = A.mask[bi];
  hv[tid] = A.hV_in[(size_t)bi*DH + tid];
  if (tid < NK){ int j = A.eidx[(size_t)bi*NK+tid]; nbS[tid]=j; mattS[tid]=mi*A.mask[b*NN+j]; }
  __syncthreads();
  for (int idx=tid; idx<NK*384; idx+=128){
    int r = idx/384, c = idx - r*384;
    float v;
    if (c < 128)      v = hv[c];
    else if (c < 256) v = A.hE[((size_t)bi*NK + r)*DH + (c-128)];
    else              v = A.hV_in[((size_t)b*NN + nbS[r])*DH + (c-256)];
    A1[r*AS1 + c] = (_Float16)v;
  }
  __syncthreads();
  for (int nt=2*w; nt<2*w+2; ++nt)
    for (int mt=0; mt<3; ++mt)
      store_act_gelu(A2, AS2, mt, nt, lane, gemm_tile(A1, AS1, A.W1t, 384, mt, nt, lane), A.b1);
  __syncthreads();
  _Float16* A3 = A1;
  for (int nt=2*w; nt<2*w+2; ++nt)
    for (int mt=0; mt<3; ++mt)
      store_act_gelu(A3, AS2, mt, nt, lane, gemm_tile(A2, AS2, A.W2t, 128, mt, nt, lane), A.b2);
  __syncthreads();
  float* fscr = (float*)A2;                 // A2 free after this barrier
  float* dh = fscr; float* h1 = fscr+128; float* red = fscr+256; float* ffn = fscr+384;
  for (int nt=2*w; nt<2*w+2; ++nt){
    float part = 0.f;
    for (int mt=0; mt<3; ++mt){
      v8f acc = gemm_tile(A3, AS2, A.W3t, 128, mt, nt, lane);
      int n = nt*16 + (lane&15), m0 = mt*16 + ((lane>>4)<<3);
      float bv = A.b3[n];
#pragma unroll
      for (int v=0; v<8; ++v) part += (acc[v]+bv)*mattS[m0+v];
    }
    part += __shfl_xor(part, 16, 32);
    if (lane < 16) dh[nt*16 + lane] = part;
  }
  __syncthreads();
  float an = A.a_node[0];
  float pre = an*(dh[tid]*(1.0f/SCALE_)) + hv[tid];
  float mu = blk_sum128(pre, red)*(1.0f/128.f);
  float dd = pre - mu;
  float var = blk_sum128(dd*dd, red)*(1.0f/128.f);
  float h1v = A.n1g[tid]*dd*rsqrtf(var+LN_EPS_) + A.n1b[tid];
  h1[tid] = h1v;
  __syncthreads();
  for (int o=tid; o<512; o+=128){
    float s = A.dinb[o];
    for (int k=0;k<128;++k) s += h1[k]*A.din[(size_t)k*512 + o];
    ffn[o] = geluf(s);
  }
  __syncthreads();
  float s2 = A.doutb[tid];
  for (int k=0;k<512;++k) s2 += ffn[k]*A.dout[(size_t)k*128 + tid];
  float pre2 = an*s2 + h1v;
  mu = blk_sum128(pre2, red)*(1.0f/128.f);
  dd = pre2 - mu;
  var = blk_sum128(dd*dd, red)*(1.0f/128.f);
  float outv = A.n2g[tid]*dd*rsqrtf(var+LN_EPS_) + A.n2b[tid];
  A.hV_out[(size_t)bi*DH + tid] = mi*outv;
}

// ---------------- encoder edge update ----------------
struct EncEdgeArgs {
  const float* hV; float* hE; const int* eidx;
  const _Float16 *W1t,*W2t,*W3t;
  const float *b1,*b2,*b3;
  const float *n3g,*n3b; const float* a_pair;
};

__global__ __launch_bounds__(128) void k_enc_edge(EncEdgeArgs A){
  constexpr int AS1 = 400, AS2 = 144;
  __shared__ _Float16 A1[NK*AS1];
  __shared__ _Float16 A2[NK*AS2];
  __shared__ int nbS[NK];
  __shared__ float hv[128];
  int bi = blockIdx.x, tid = threadIdx.x, lane = tid&31, w = tid>>5;
  int b = bi/NN;
  hv[tid] = A.hV[(size_t)bi*DH + tid];
  if (tid < NK) nbS[tid] = A.eidx[(size_t)bi*NK + tid];
  __syncthreads();
  for (int idx=tid; idx<NK*384; idx+=128){
    int r = idx/384, c = idx - r*384;
    float v;
    if (c < 128)      v = hv[c];
    else if (c < 256) v = A.hE[((size_t)bi*NK + r)*DH + (c-128)];
    else              v = A.hV[((size_t)b*NN + nbS[r])*DH + (c-256)];
    A1[r*AS1 + c] = (_Float16)v;
  }
  __syncthreads();
  for (int nt=2*w; nt<2*w+2; ++nt)
    for (int mt=0; mt<3; ++mt)
      store_act_gelu(A2, AS2, mt, nt, lane, gemm_tile(A1, AS1, A.W1t, 384, mt, nt, lane), A.b1);
  __syncthreads();
  _Float16* A3 = A1;
  for (int nt=2*w; nt<2*w+2; ++nt)
    for (int mt=0; mt<3; ++mt)
      store_act_gelu(A3, AS2, mt, nt, lane, gemm_tile(A2, AS2, A.W2t, 128, mt, nt, lane), A.b2);
  __syncthreads();
  float* preB = (float*)(A1 + NK*AS2);      // beyond A3's footprint: 48*128 f32 fits exactly
  float ap = A.a_pair[0];
  for (int nt=2*w; nt<2*w+2; ++nt)
    for (int mt=0; mt<3; ++mt){
      v8f acc = gemm_tile(A3, AS2, A.W3t, 128, mt, nt, lane);
      int n = nt*16 + (lane&15), m0 = mt*16 + ((lane>>4)<<3);
      float bv = A.b3[n];
#pragma unroll
      for (int v=0; v<8; ++v){
        int m = m0+v;
        preB[m*128 + n] = ap*(acc[v]+bv) + A.hE[((size_t)bi*NK + m)*DH + n];
      }
    }
  __syncthreads();
  for (int r = w*12; r < w*12+12; ++r){
    float vv[4]; float s = 0.f;
#pragma unroll
    for (int q=0;q<4;++q){ vv[q]=preB[r*128 + lane + 32*q]; s += vv[q]; }
    for (int m=16;m>0;m>>=1) s += __shfl_xor(s,m,32);
    float mu = s*(1.f/128.f), s2 = 0.f;
#pragma unroll
    for (int q=0;q<4;++q){ float d=vv[q]-mu; s2 += d*d; }
    for (int m=16;m>0;m>>=1) s2 += __shfl_xor(s2,m,32);
    float rin = rsqrtf(s2*(1.f/128.f)+LN_EPS_);
#pragma unroll
    for (int q=0;q<4;++q){ int c=lane+32*q;
      A.hE[((size_t)bi*NK + r)*DH + c] = A.n3g[c]*(vv[q]-mu)*rin + A.n3b[c]; }
  }
}

// ---------------- decoder layer ----------------
struct DecArgs {
  const float* hVd_in; float* hVd_out;
  const float* hE; const float* hVenc; const int* eidx; const float* mask;
  const int* invp; const int* tok; const float* emb;
  const _Float16 *W1t,*W2t,*W3t;
  const float *b1,*b2,*b3;
  const float *din,*dinb,*dout,*doutb;
  const float *n1g,*n1b,*n2g,*n2b;
  const float* a_node;
  float* esv_out; int write_esv;
};

__global__ __launch_bounds__(128) void k_dec(DecArgs A){
  constexpr int AS1 = 512, AS2 = 144;
  __shared__ _Float16 A1[NK*AS1];
  __shared__ _Float16 A2[NK*AS2];
  __shared__ int   nbS[NK];
  __shared__ float bwS[NK], fwS[NK];
  __shared__ float hv[128];
  int bi = blockIdx.x, tid = threadIdx.x, lane = tid&31, w = tid>>5;
  int b = bi/NN;
  float mi = A.mask[bi];
  hv[tid] = A.hVd_in[(size_t)bi*DH + tid];
  if (tid < NK){
    int j = A.eidx[(size_t)bi*NK + tid]; nbS[tid]=j;
    float att = (A.invp[bi] > A.invp[b*NN + j]) ? 1.f : 0.f;
    bwS[tid] = mi*att; fwS[tid] = mi*(1.f-att);
  }
  __syncthreads();
  for (int idx=tid; idx<NK*512; idx+=128){
    int r = idx>>9, c = idx&511; int j = nbS[r];
    float v;
    if (c < 128)      v = hv[c];
    else if (c < 256) v = mi * A.hE[((size_t)bi*NK + r)*DH + (c-128)];
    else if (c < 384) v = bwS[r] * A.emb[(size_t)A.tok[b*NN+j]*DH + (c-256)];
    else              v = bwS[r]*A.hVd_in[((size_t)b*NN+j)*DH + (c-384)]
                        + fwS[r]*A.hVenc[((size_t)b*NN+j)*DH + (c-384)];
    A1[r*AS1 + c] = (_Float16)v;
    if (A.write_esv && c >= 128) A.esv_out[((size_t)bi*NK + r)*384 + (c-128)] = v;
  }
  __syncthreads();
  for (int nt=2*w; nt<2*w+2; ++nt)
    for (int mt=0; mt<3; ++mt)
      store_act_gelu(A2, AS2, mt, nt, lane, gemm_tile(A1, AS1, A.W1t, 512, mt, nt, lane), A.b1);
  __syncthreads();
  _Float16* A3 = A1;
  for (int nt=2*w; nt<2*w+2; ++nt)
    for (int mt=0; mt<3; ++mt)
      store_act_gelu(A3, AS2, mt, nt, lane, gemm_tile(A2, AS2, A.W2t, 128, mt, nt, lane), A.b2);
  __syncthreads();
  float* fscr = (float*)A2;
  float* dh = fscr; float* h1 = fscr+128; float* red = fscr+256; float* ffn = fscr+384;
  for (int nt=2*w; nt<2*w+2; ++nt){
    float part = 0.f;
    for (int mt=0; mt<3; ++mt){
      v8f acc = gemm_tile(A3, AS2, A.W3t, 128, mt, nt, lane);
      int n = nt*16 + (lane&15);
      float bv = A.b3[n];
#pragma unroll
      for (int v=0; v<8; ++v) part += acc[v] + bv;
    }
    part += __shfl_xor(part, 16, 32);
    if (lane < 16) dh[nt*16 + lane] = part;
  }
  __syncthreads();
  float an = A.a_node[0];
  float pre = an*(dh[tid]*(1.0f/SCALE_)) + hv[tid];
  float mu = blk_sum128(pre, red)*(1.0f/128.f);
  float dd = pre - mu;
  float var = blk_sum128(dd*dd, red)*(1.0f/128.f);
  float h1v = A.n1g[tid]*dd*rsqrtf(var+LN_EPS_) + A.n1b[tid];
  h1[tid] = h1v;
  __syncthreads();
  for (int o=tid; o<512; o+=128){
    float s = A.dinb[o];
    for (int k=0;k<128;++k) s += h1[k]*A.din[(size_t)k*512 + o];
    ffn[o] = geluf(s);
  }
  __syncthreads();
  float s2 = A.doutb[tid];
  for (int k=0;k<512;++k) s2 += ffn[k]*A.dout[(size_t)k*128 + tid];
  float pre2 = an*s2 + h1v;
  mu = blk_sum128(pre2, red)*(1.0f/128.f);
  dd = pre2 - mu;
  var = blk_sum128(dd*dd, red)*(1.0f/128.f);
  float outv = A.n2g[tid]*dd*rsqrtf(var+LN_EPS_) + A.n2b[tid];
  A.hVd_out[(size_t)bi*DH + tid] = mi*outv;
}

// ---------------- host launcher ----------------
extern "C" void kernel_launch(void* const* d_in, const int* in_sizes, int n_in,
                              void* d_out, int out_size, void* d_ws, size_t ws_size,
                              hipStream_t stream){
  (void)in_sizes; (void)n_in; (void)out_size; (void)ws_size;
  const float* node_feats = (const float*)d_in[0];
  const float* pair_feats = (const float*)d_in[1];
  const float* res_mask   = (const float*)d_in[2];
  const float* ca         = (const float*)d_in[3];
  const int*   seq_tok    = (const int*)d_in[5];
  const int*   dec_order  = (const int*)d_in[6];
  // params pytree (alphabetical dict keys): dec[3] (15 each), enc[3] (24 each), then globals
  const int PDEC = 7, PENC = 7+45, PGL = 7+45+72;   // PGL = 124
  auto F = [&](int i){ return (const float*)d_in[i]; };

  uint8_t* ws = (uint8_t*)d_ws;
  size_t off = 0;
  auto carve = [&](size_t bytes)->void*{ void* p = ws + off; off = (off + bytes + 255) & ~(size_t)255; return p; };
  int*   eidx = (int*)carve((size_t)BB*NN*NK*4);
  int*   invb = (int*)carve((size_t)BB*NN*4);
  float* hE   = (float*)carve((size_t)BB*NN*NK*DH*4);
  float* hva  = (float*)carve((size_t)BB*NN*DH*4);
  float* hvb  = (float*)carve((size_t)BB*NN*DH*4);
  float* hvc  = (float*)carve((size_t)BB*NN*DH*4);
  _Float16* wh = (_Float16*)carve((size_t)900000*2);

  _Float16* pairWt = wh;
  _Float16* hp = wh + 128*128;
  _Float16 *eW1t[3],*eW11t[3],*eW12t[3],*eW13t[3],*eW2t[3],*eW3t[3];
  _Float16 *dW1t[3],*dW2t[3],*dW3t[3];
  for (int l=0;l<3;++l){
    eW1t[l]=hp;  hp+=384*128;
    eW11t[l]=hp; hp+=384*128;
    eW12t[l]=hp; hp+=128*128;
    eW13t[l]=hp; hp+=128*128;
    eW2t[l]=hp;  hp+=128*128;
    eW3t[l]=hp;  hp+=128*128;
  }
  for (int l=0;l<3;++l){ dW1t[l]=hp; hp+=512*128; dW2t[l]=hp; hp+=128*128; dW3t[l]=hp; hp+=128*128; }

  auto w2h = [&](const float* s, _Float16* d, int Kd, int Nd){
    int tot = Kd*Nd;
    k_w2h<<<(tot+255)/256, 256, 0, stream>>>(s, d, Kd, Nd);
  };
  w2h(F(PGL+4), pairWt, 128, 128);                       // pair_W
  for (int l=0;l<3;++l){
    int e = PENC + 24*l;
    w2h(F(e+0), eW1t[l],  384, 128);
    w2h(F(e+1), eW11t[l], 384, 128);
    w2h(F(e+2), eW12t[l], 128, 128);
    w2h(F(e+3), eW13t[l], 128, 128);
    w2h(F(e+4), eW2t[l],  128, 128);
    w2h(F(e+5), eW3t[l],  128, 128);
  }
  for (int l=0;l<3;++l){
    int e = PDEC + 15*l;
    w2h(F(e+0), dW1t[l], 512, 128);
    w2h(F(e+1), dW2t[l], 128, 128);
    w2h(F(e+2), dW3t[l], 128, 128);
  }

  k_inv<<<(BB*NN+255)/256, 256, 0, stream>>>(dec_order, invb);
  k_knn<<<BB*NN, 256, 0, stream>>>(ca, res_mask, eidx);
  k_node_embed<<<BB*NN, 128, 0, stream>>>(node_feats, F(PGL+3), F(PGL+2), F(PGL+0), F(PGL+1), hva);
  k_edge_embed<<<BB*NN, 128, 0, stream>>>(pair_feats, eidx, F(PGL+7), F(PGL+6), pairWt, F(PGL+5), hE);

  float* hin = hva; float* hout = hvb;
  for (int l=0; l<3; ++l){
    int e = PENC + 24*l;
    EncNodeArgs na{hin, hout, hE, eidx, res_mask,
                   eW1t[l], eW2t[l], eW3t[l],
                   F(e+8), F(e+12), F(e+13),
                   F(e+14), F(e+15), F(e+16), F(e+17),
                   F(e+19), F(e+18), F(e+21), F(e+20),
                   F(e+6)};
    k_enc_node<<<BB*NN, 128, 0, stream>>>(na);
    EncEdgeArgs ea{hout, hE, eidx,
                   eW11t[l], eW12t[l], eW13t[l],
                   F(e+9), F(e+10), F(e+11),
                   F(e+23), F(e+22), F(e+7)};
    k_enc_edge<<<BB*NN, 128, 0, stream>>>(ea);
    float* t = hin; hin = hout; hout = t;
  }
  const float* hVenc = hin;                  // == hvb after 3 swaps
  float* dout_f = (float*)d_out;
  float* esv = dout_f + (size_t)BB*NN*DH;

  const float* dvin = hVenc;
  float* buf_next[3] = { hvc, hva, dout_f };
  for (int l=0; l<3; ++l){
    int e = PDEC + 15*l;
    DecArgs da{dvin, buf_next[l], hE, hVenc, eidx, res_mask,
               invb, seq_tok, F(PGL+8),
               dW1t[l], dW2t[l], dW3t[l],
               F(e+4), F(e+5), F(e+6),
               F(e+7), F(e+8), F(e+9), F(e+10),
               F(e+12), F(e+11), F(e+14), F(e+13),
               F(e+3),
               esv, (l==2) ? 1 : 0};
    k_dec<<<BB*NN, 128, 0, stream>>>(da);
    dvin = buf_next[l];
  }
}
